// CausalSelfAttention_60284160966732
// MI455X (gfx1250) — compile-verified, matrix operand order checked by reading, not run on hardware
//
#include <hip/hip_runtime.h>
#include <cstdint>
#include <cstddef>

// ---------------------------------------------------------------------------
// CDNA5 (gfx1250) causal self-attention: bf16 WMMA + TDM async tensor loads.
// ---------------------------------------------------------------------------

typedef __attribute__((ext_vector_type(16))) __bf16 v16bf;
typedef __attribute__((ext_vector_type(8)))  float  v8f;
typedef __attribute__((ext_vector_type(4)))  unsigned int v4u;
typedef __attribute__((ext_vector_type(8)))  unsigned int v8u;

#define B_   8
#define T_   1024
#define C_   1024
#define H_   16
#define D_   64
#define C3_  3072

// float -> bf16 bits, round-to-nearest-even
__device__ __forceinline__ unsigned short f2bf(float f) {
  unsigned u = __float_as_uint(f);
  u += 0x7FFFu + ((u >> 16) & 1u);
  return (unsigned short)(u >> 16);
}

// Load one 16x32 bf16 A/B fragment (ISA 7.12.2 16-bit A layout, wave32):
// lane in [0,15]: row M=lane, K 0..7/16..23 ; lane>=16: K 8..15/24..31.
// Row-major source with K contiguous -> two 16B chunks per lane.
__device__ __forceinline__ v16bf ldfrag(const unsigned short* rowptr, int kk, int hh) {
  const uint4* p = reinterpret_cast<const uint4*>(rowptr);
  union { uint4 u[2]; v16bf v; } f;
  f.u[0] = p[(kk >> 3) + hh];
  f.u[1] = p[(kk >> 3) + hh + 2];
  return f.v;
}

// ---------------------------------------------------------------------------
// Tensor Data Mover: async-load a [64 rows x 32 K-elems] bf16 tile of a
// row-major [rows_total x K] matrix into LDS (contiguous 64x32 layout).
// D# per CDNA5 ISA:
//   group0: count=1 | lds_addr | global_addr | type=2
//   group1: data_size=2B, tensor_dim0=K, tensor_dim1=rows, tile 32x64,
//           tensor_dim0_stride=K
// Descriptors are wave-uniform -> SGPRs; EXEC is ignored by TDM.
// ---------------------------------------------------------------------------
__device__ __forceinline__ void tdm_load_tile_b(const unsigned short* gptr,
                                                unsigned lds_byte,
                                                int rows_total, int K) {
  unsigned long long ga = (unsigned long long)(uintptr_t)gptr;
  v4u g0;
  g0.x = 1u;                                   // count=1 (valid), user mode
  g0.y = lds_byte;                             // lds_addr (bytes)
  g0.z = (unsigned)(ga & 0xFFFFFFFFu);         // global_addr[31:0]
  g0.w = (unsigned)((ga >> 32) & 0x01FFFFFFu)  // global_addr[56:32]
       | (2u << 30);                           // type=2 ("image")
  v8u g1;
  g1.s0 = (1u << 16);                                        // data_size=2B
  g1.s1 = ((unsigned)K & 0xFFFFu) << 16;                     // tensor_dim0 lo
  g1.s2 = (((unsigned)K >> 16) & 0xFFFFu)                    // tensor_dim0 hi
        | (((unsigned)rows_total & 0xFFFFu) << 16);          // tensor_dim1 lo
  g1.s3 = (((unsigned)rows_total >> 16) & 0xFFFFu)           // tensor_dim1 hi
        | (32u << 16);                                       // tile_dim0 = 32
  g1.s4 = 64u;                                               // tile_dim1 = 64
  g1.s5 = (unsigned)K;                                       // dim0_stride lo
  g1.s6 = 0u;
  g1.s7 = 0u;
  v4u g2 = (v4u)(0u);   // 2-D tile: dims 2..4 unused (zero descriptors)
  v4u g3 = (v4u)(0u);
  asm volatile("tensor_load_to_lds %0, %1, %2, %3"
               :: "s"(g0), "s"(g1), "s"(g2), "s"(g3)
               : "memory");
}

// ---------------------------------------------------------------------------
// fp32 -> bf16 elementwise
// ---------------------------------------------------------------------------
__global__ void cvt_f32_bf16(const float* __restrict__ in,
                             unsigned short* __restrict__ out, int n) {
  int i = blockIdx.x * blockDim.x + threadIdx.x;
  int stride = gridDim.x * blockDim.x;
  for (; i < n; i += stride) out[i] = f2bf(in[i]);
}

// ---------------------------------------------------------------------------
// NT GEMM:  C[M,N] = A[M,K] (row-major) * B[N,K]^T (row-major, K contiguous)
// Block = 4 waves = 64x64 C tile. The 64x32 B panel is TDM-DMA'd into LDS
// (double-buffered, overlapped with compute) and shared by all 4 waves;
// each wave computes a 16x64 strip (1 A-frag + 4 LDS B-frags -> 4 WMMAs /step).
// ---------------------------------------------------------------------------
template <bool F32OUT>
__global__ void gemm_nt_wmma(const unsigned short* __restrict__ A,
                             const unsigned short* __restrict__ Bw,
                             const float* __restrict__ bias,
                             unsigned short* __restrict__ Cb,
                             float* __restrict__ Cf,
                             int M, int N, int K) {
  __shared__ __align__(16) unsigned short sB[2][64][32];   // 8KB double buffer

  int w    = threadIdx.x >> 5;
  int lane = threadIdx.x & 31;
  int ln   = lane & 15;
  int hh   = lane >> 4;

  int tilesN = N >> 6;
  int tm0 = (blockIdx.x / tilesN) << 6;
  int tn0 = (blockIdx.x % tilesN) << 6;

  const unsigned short* arow = A + (size_t)(tm0 + 16 * w + ln) * (size_t)K;

  v8f acc0 = {}, acc1 = {}, acc2 = {}, acc3 = {};

  // prologue: DMA tile 0
  if (w == 0) {
    tdm_load_tile_b(Bw + (size_t)tn0 * K, (unsigned)(size_t)&sB[0][0][0], N, K);
    __builtin_amdgcn_s_wait_tensorcnt(0);
  }
  __syncthreads();

  int nk = K >> 5;
  for (int i = 0; i < nk; ++i) {
    int kk = i << 5;
    // kick off next panel's DMA while we compute on the current one
    if (w == 0 && i + 1 < nk) {
      tdm_load_tile_b(Bw + (size_t)tn0 * K + (kk + 32),
                      (unsigned)(size_t)&sB[(i + 1) & 1][0][0], N, K);
    }
    __builtin_prefetch(arow + kk + 512, 0, 1);   // global_prefetch_b8
    v16bf fa = ldfrag(arow, kk, hh);
    const unsigned short* bb = &sB[i & 1][0][0];
    v16bf fb0 = ldfrag(bb + (0  + ln) * 32, 0, hh);
    v16bf fb1 = ldfrag(bb + (16 + ln) * 32, 0, hh);
    v16bf fb2 = ldfrag(bb + (32 + ln) * 32, 0, hh);
    v16bf fb3 = ldfrag(bb + (48 + ln) * 32, 0, hh);
    acc0 = __builtin_amdgcn_wmma_f32_16x16x32_bf16(false, fa, false, fb0, (short)0, acc0, false, false);
    acc1 = __builtin_amdgcn_wmma_f32_16x16x32_bf16(false, fa, false, fb1, (short)0, acc1, false, false);
    acc2 = __builtin_amdgcn_wmma_f32_16x16x32_bf16(false, fa, false, fb2, (short)0, acc2, false, false);
    acc3 = __builtin_amdgcn_wmma_f32_16x16x32_bf16(false, fa, false, fb3, (short)0, acc3, false, false);
    __syncthreads();                                  // everyone done reading buf i&1
    if (w == 0 && i + 1 < nk) __builtin_amdgcn_s_wait_tensorcnt(0);
    __syncthreads();                                  // buf (i+1)&1 published
  }

#pragma unroll
  for (int r = 0; r < 8; ++r) {
    size_t row = (size_t)(tm0 + 16 * w + r + 8 * hh);
    int    c0  = tn0 + ln;
    float  v0 = acc0[r], v1 = acc1[r], v2 = acc2[r], v3 = acc3[r];
    if constexpr (F32OUT) {
      Cf[row * (size_t)N + c0 +  0] = v0 + bias[c0 +  0];
      Cf[row * (size_t)N + c0 + 16] = v1 + bias[c0 + 16];
      Cf[row * (size_t)N + c0 + 32] = v2 + bias[c0 + 32];
      Cf[row * (size_t)N + c0 + 48] = v3 + bias[c0 + 48];
    } else {
      Cb[row * (size_t)N + c0 +  0] = f2bf(v0);
      Cb[row * (size_t)N + c0 + 16] = f2bf(v1);
      Cb[row * (size_t)N + c0 + 32] = f2bf(v2);
      Cb[row * (size_t)N + c0 + 48] = f2bf(v3);
    }
  }
}

// ---------------------------------------------------------------------------
// V slice of qkv -> vT[B,H,D,T]  (so PV-GEMM's B operand is NT-layout)
// ---------------------------------------------------------------------------
__global__ void transpose_v_kernel(const unsigned short* __restrict__ qkv,
                                   unsigned short* __restrict__ vT) {
  int idx = blockIdx.x * blockDim.x + threadIdx.x;   // B*H*D*T = 8M
  int t  = idx & (T_ - 1);
  int d  = (idx >> 10) & (D_ - 1);
  int hb = idx >> 16;             // b*H + h
  int h  = hb & (H_ - 1);
  int b  = hb >> 4;
  vT[idx] = qkv[((size_t)(b * T_ + t)) * C3_ + 2 * C_ + h * D_ + d];
}

// ---------------------------------------------------------------------------
// Flash attention: one wave per (b, h, 16-row Q tile).
// KV streamed in 32-row chunks; online softmax; P bounced through LDS to
// re-layout C-fragment -> A-fragment; O accumulated as 4 fp32 WMMA tiles.
// qkv split order (per reference): [k | q | v] along last dim.
// ---------------------------------------------------------------------------
__global__ void flash_attn_kernel(const unsigned short* __restrict__ qkv,
                                  const unsigned short* __restrict__ vT,
                                  unsigned short* __restrict__ y) {
  __shared__ __align__(16) unsigned short pbuf[4][16][32];  // per-wave 1KB P tile

  int w    = threadIdx.x >> 5;
  int lane = threadIdx.x & 31;
  int ln   = lane & 15;
  int hh   = lane >> 4;

  int wid = blockIdx.x * 4 + w;
  int qt  = wid & 63;          // q tile within (b,h)
  int h   = (wid >> 6) & 15;
  int b   = wid >> 10;
  int q0  = qt << 4;

  const unsigned short* kbase = qkv + (size_t)b * T_ * C3_ + h * D_;        // keys
  const unsigned short* qbase = kbase + C_;                                  // queries
  const unsigned short* vtb   = vT + ((size_t)(b * H_ + h) * D_) * (size_t)T_;

  // Q fragments for this tile (head dim K=64 -> two 32-wide fragments)
  const unsigned short* qrow = qbase + (size_t)(q0 + ln) * C3_;
  v16bf qf0 = ldfrag(qrow, 0, hh);
  v16bf qf1 = ldfrag(qrow, 32, hh);

  float mrow[8], lrow[8];
  v8f o0 = {}, o1 = {}, o2 = {}, o3 = {};
#pragma unroll
  for (int r = 0; r < 8; ++r) { mrow[r] = -1e30f; lrow[r] = 0.0f; }

  int nch = (q0 + 47) >> 5;  // 32-row KV chunks covering [0, q0+16)
  for (int ch = 0; ch < nch; ++ch) {
    int kst = ch << 5;
    const unsigned short* kr0 = kbase + (size_t)(kst + ln) * C3_;
    const unsigned short* kr1 = kbase + (size_t)(kst + 16 + ln) * C3_;

    // S = Q * K^T  (two 16x16 n-tiles per chunk, K-dim 64 = 2x32)
    v8f s0 = {}, s1 = {};
    s0 = __builtin_amdgcn_wmma_f32_16x16x32_bf16(false, qf0, false, ldfrag(kr0, 0, hh),  (short)0, s0, false, false);
    s0 = __builtin_amdgcn_wmma_f32_16x16x32_bf16(false, qf1, false, ldfrag(kr0, 32, hh), (short)0, s0, false, false);
    s1 = __builtin_amdgcn_wmma_f32_16x16x32_bf16(false, qf0, false, ldfrag(kr1, 0, hh),  (short)0, s1, false, false);
    s1 = __builtin_amdgcn_wmma_f32_16x16x32_bf16(false, qf1, false, ldfrag(kr1, 32, hh), (short)0, s1, false, false);

    // scale + causal mask + online softmax (rows r+8*hh, kv col = n-tile*16+ln)
#pragma unroll
    for (int r = 0; r < 8; ++r) {
      int   qg = q0 + r + 8 * hh;
      float a  = s0[r] * 0.125f;               // 1/sqrt(64)
      float c  = s1[r] * 0.125f;
      if (kst + ln > qg)      a = -1e30f;
      if (kst + 16 + ln > qg) c = -1e30f;

      float mc = fmaxf(a, c);
#pragma unroll
      for (int off = 1; off < 16; off <<= 1) mc = fmaxf(mc, __shfl_xor(mc, off, 32));
      float mn   = fmaxf(mrow[r], mc);
      float corr = __expf(mrow[r] - mn);
      mrow[r]    = mn;

      float p0 = __expf(a - mn);
      float p1 = __expf(c - mn);
      float ps = p0 + p1;
#pragma unroll
      for (int off = 1; off < 16; off <<= 1) ps += __shfl_xor(ps, off, 32);
      lrow[r] = lrow[r] * corr + ps;

      o0[r] *= corr; o1[r] *= corr; o2[r] *= corr; o3[r] *= corr;

      pbuf[w][r + 8 * hh][ln]      = f2bf(p0);
      pbuf[w][r + 8 * hh][16 + ln] = f2bf(p1);
    }

    // Re-read P in A-fragment layout (same wave: LDS ops in-order, no barrier)
    asm volatile("" ::: "memory");
    const uint4* pr = reinterpret_cast<const uint4*>(&pbuf[w][ln][0]);
    union { uint4 u[2]; v16bf v; } pf;
    pf.u[0] = pr[hh];
    pf.u[1] = pr[hh + 2];
    asm volatile("" ::: "memory");

    // O += P * V  (4 n-tiles of head dim, V^T rows are K-contiguous)
    const unsigned short* vr0 = vtb + (size_t)(0  + ln) * T_ + kst;
    const unsigned short* vr1 = vtb + (size_t)(16 + ln) * T_ + kst;
    const unsigned short* vr2 = vtb + (size_t)(32 + ln) * T_ + kst;
    const unsigned short* vr3 = vtb + (size_t)(48 + ln) * T_ + kst;
    o0 = __builtin_amdgcn_wmma_f32_16x16x32_bf16(false, pf.v, false, ldfrag(vr0, 0, hh), (short)0, o0, false, false);
    o1 = __builtin_amdgcn_wmma_f32_16x16x32_bf16(false, pf.v, false, ldfrag(vr1, 0, hh), (short)0, o1, false, false);
    o2 = __builtin_amdgcn_wmma_f32_16x16x32_bf16(false, pf.v, false, ldfrag(vr2, 0, hh), (short)0, o2, false, false);
    o3 = __builtin_amdgcn_wmma_f32_16x16x32_bf16(false, pf.v, false, ldfrag(vr3, 0, hh), (short)0, o3, false, false);
  }

  // normalize + store y[b, q, h*64 + d] as bf16 (input to output-proj GEMM)
#pragma unroll
  for (int r = 0; r < 8; ++r) {
    float  inv = 1.0f / lrow[r];
    size_t row = (size_t)(b * T_ + q0 + r + 8 * hh);
    size_t col = (size_t)(h * D_ + ln);
    y[row * C_ + col +  0] = f2bf(o0[r] * inv);
    y[row * C_ + col + 16] = f2bf(o1[r] * inv);
    y[row * C_ + col + 32] = f2bf(o2[r] * inv);
    y[row * C_ + col + 48] = f2bf(o3[r] * inv);
  }
}

// ---------------------------------------------------------------------------
// Host launcher
// ---------------------------------------------------------------------------
extern "C" void kernel_launch(void* const* d_in, const int* in_sizes, int n_in,
                              void* d_out, int out_size, void* d_ws, size_t ws_size,
                              hipStream_t stream) {
  (void)in_sizes; (void)n_in; (void)out_size; (void)ws_size;

  const float* x      = (const float*)d_in[0];
  const float* w_attn = (const float*)d_in[1];
  const float* w_proj = (const float*)d_in[2];
  const float* b_proj = (const float*)d_in[3];
  float*       out    = (float*)d_out;

  char* ws = (char*)d_ws;
  // bf16 workspace layout (bytes):
  unsigned short* xb   = (unsigned short*)(ws + 0);         // 8M  elems, 16MB
  unsigned short* wab  = (unsigned short*)(ws + 16777216);  // 3M  elems,  6MB
  unsigned short* wpb  = (unsigned short*)(ws + 23068672);  // 1M  elems,  2MB
  unsigned short* qkvb = (unsigned short*)(ws + 25165824);  // 24M elems, 48MB
  unsigned short* vTb  = (unsigned short*)(ws + 75497472);  // 8M  elems, 16MB
  unsigned short* yb   = (unsigned short*)(ws + 92274688);  // 8M  elems, 16MB

  // 1) fp32 -> bf16
  cvt_f32_bf16<<<4096, 256, 0, stream>>>(x,      xb,  B_ * T_ * C_);
  cvt_f32_bf16<<<4096, 256, 0, stream>>>(w_attn, wab, 3 * C_ * C_);
  cvt_f32_bf16<<<2048, 256, 0, stream>>>(w_proj, wpb, C_ * C_);

  // 2) qkv = x @ w_attn^T   (M=8192, N=3072, K=1024): 128x48 block tiles of 64x64
  gemm_nt_wmma<false><<<6144, 128, 0, stream>>>(xb, wab, nullptr, qkvb, nullptr,
                                                B_ * T_, C3_, C_);

  // 3) V -> V^T [B,H,D,T]
  transpose_v_kernel<<<32768, 256, 0, stream>>>(qkvb, vTb);

  // 4) flash attention: 8*16*64 = 8192 waves, 4 waves/block
  flash_attn_kernel<<<2048, 128, 0, stream>>>(qkvb, vTb, yb);

  // 5) out = y @ w_proj^T + b   (M=8192, N=1024, K=1024): 128x16 block tiles
  gemm_nt_wmma<true><<<2048, 128, 0, stream>>>(yb, wpb, b_proj, nullptr, out,
                                               B_ * T_, C_, C_);
}
